// VectorQuantizer_83141976916941
// MI455X (gfx1250) — matrix operand order repaired and checked by reading, not compiled, not run on hardware
//
#include <hip/hip_runtime.h>
#include <hip/hip_bf16.h>

typedef __attribute__((ext_vector_type(16))) __bf16        v16bf;
typedef __attribute__((ext_vector_type(8)))  float         v8f;
typedef __attribute__((ext_vector_type(8)))  unsigned int  v8u;
typedef __attribute__((ext_vector_type(4)))  unsigned int  u32x4;
typedef __attribute__((ext_vector_type(8)))  int           i32x8;
typedef __attribute__((ext_vector_type(4)))  int           i32x4;

#define NPIX   32768          // 32*32*32 pixels
#define NE     2048           // codes
#define KDIM   256            // channels
#define KPAD   288            // K padded: 256 data + 1 norm + 31 zero -> 9 chunks of 32
#define NKC    9              // K-chunks of 32
#define NCHUNK 128            // 2048 / 16 code chunks
#define PXT    128            // pixels per workgroup (8 waves x 16)
#define ZQ_N   (NPIX*KDIM)    // 8388608 z_q elements
#define CHUNK_ELTS (16*KPAD)  // 4608 ushorts per (chunk, hi/lo) tile
#define EARR_ELTS  ((unsigned)NE*KPAD)   // 589824 = 0x90000 elements per array

// ---------- helpers: round-to-nearest-even bf16 + hi/lo split ----------
__device__ __forceinline__ unsigned bf16_rne(float x) {
  unsigned u = __float_as_uint(x);
  return (u + 0x7FFFu + ((u >> 16) & 1u)) >> 16;
}
__device__ __forceinline__ void split2(float x, unsigned &h, unsigned &l) {
  h = bf16_rne(x);
  float hf = __uint_as_float(h << 16);
  l = bf16_rne(x - hf);
}

// ---------- TDM: DMA one (16 x KPAD) ushort tile from global into LDS ----------
// D# per CDNA5 ISA ch.8: group0 = {flags, lds_addr, global_addr, type=2},
// group1 = dims/strides (data_size=2B, tile_dim0=4608, tile_dim1=1),
// groups2/3 = degenerate higher dims (=1). Tracked by TENSORcnt.
// This toolchain exposes the 6-arg builtin: (u32x4, i32x8, i32x4, i32x4, i32x8, cpol).
__device__ __forceinline__ void tdm_load_chunk(const unsigned short* gsrc,
                                               unsigned lds_byte_off) {
  unsigned long long ga = (unsigned long long)(uintptr_t)gsrc;
  u32x4 g0;
  g0[0] = 1u;                                        // count=1, user-mode, no gather
  g0[1] = lds_byte_off;                              // LDS dest byte offset (base 0: single shared object)
  g0[2] = (unsigned)(ga & 0xFFFFFFFFu);              // global_addr[31:0]
  g0[3] = (unsigned)((ga >> 32) & 0x01FFFFFFu) | 0x80000000u;  // global_addr[56:32] | type=2
  i32x8 g1;
  g1[0] = 0x00010000;        // wg_mask=0 (not in cluster), data_size=1 (2 bytes), no pad/iterate
  g1[1] = 0;                 // atomic_barrier=0 ; tensor_dim0[15:0] (0x90000 -> 0) << 16
  g1[2] = 0x00010009;        // tensor_dim0[31:16]=9 | tensor_dim1(=1)[15:0] << 16
  g1[3] = (int)0x12000000;   // tensor_dim1[31:16]=0 | tile_dim0 (4608=0x1200) << 16
  g1[4] = 1;                 // tile_dim1 = 1, tile_dim2 = 0 (unused)
  g1[5] = 0x00090000;        // tensor_dim0_stride[31:0] = 589824
  g1[6] = 0;                 // stride0[47:32]=0 | stride1[15:0]=0
  g1[7] = 9;                 // tensor_dim1_stride[47:16] = 0x90000 >> 16
  i32x4 g2;
  g2[0] = 1;                 // tensor_dim2 = 1
  g2[1] = 1;                 // tensor_dim3 = 1 (iterate disabled)
  g2[2] = 0x00090000;        // tensor_dim2_stride[31:0]
  g2[3] = 0;                 // stride2[47:32]=0 | tile_dim3=0 (unused)
  i32x4 g3;
  g3[0] = 0x00090000;        // tensor_dim3_stride[31:0]
  g3[1] = 0x00010000;        // stride3[47:32]=0 | tensor_dim4(=1)[15:0] << 16
  g3[2] = 0;                 // tensor_dim4[31:16]=0 | tile_dim4=0 (unused)
  g3[3] = 0;
  i32x8 g4 = {0, 0, 0, 0, 0, 0, 0, 0};  // trailing descriptor group (unused here)
  __builtin_amdgcn_tensor_load_to_lds(g0, g1, g2, g3, g4, 0);
}

// ---------- kernel 0: zero code-usage counts ----------
__global__ void vq_zero(float* __restrict__ cnt) {
  int i = blockIdx.x * 256 + threadIdx.x;
  if (i < NE) cnt[i] = 0.f;
}

// ---------- kernel 1: split codebook to bf16 hi/lo, fold ||e||^2 into padded K ----------
// one wave per code row; grid 256 x 256 threads = 2048 waves
__global__ void vq_prep(const float* __restrict__ emb,
                        unsigned short* __restrict__ ehi,
                        unsigned short* __restrict__ elo) {
  int gtid = blockIdx.x * 256 + threadIdx.x;
  int code = gtid >> 5;
  int lane = gtid & 31;
  const float* row = emb + code * KDIM;
  float v[8];
  float nrm = 0.f;
#pragma unroll
  for (int t = 0; t < 8; ++t) { v[t] = row[lane + 32 * t]; nrm += v[t] * v[t]; }
#pragma unroll
  for (int off = 16; off >= 1; off >>= 1) nrm += __shfl_xor(nrm, off, 32);
  unsigned short* hr = ehi + (size_t)code * KPAD;
  unsigned short* lr = elo + (size_t)code * KPAD;
#pragma unroll
  for (int t = 0; t < 8; ++t) {
    unsigned h, l; split2(v[t], h, l);
    hr[lane + 32 * t] = (unsigned short)h;
    lr[lane + 32 * t] = (unsigned short)l;
  }
  // pad region k = 256..287: k==256 holds ||e||^2, rest zero
  float pad = (lane == 0) ? nrm : 0.f;
  unsigned h, l; split2(pad, h, l);
  hr[KDIM + lane] = (unsigned short)h;
  lr[KDIM + lane] = (unsigned short)l;
}

// ---------- kernel 2: fused distance-WMMA + argmin + gather + loss ----------
// block = 256 threads (8 waves); wave w owns 16 pixels; all waves sweep all 2048 codes.
// Codebook chunks stream into double-buffered LDS via the Tensor Data Mover.
__global__ void __launch_bounds__(256) vq_main(
    const float* __restrict__ z, const float* __restrict__ emb,
    const unsigned short* __restrict__ ehi, const unsigned short* __restrict__ elo,
    float* __restrict__ counts, float* __restrict__ loss_part,
    float* __restrict__ out) {
  // ONLY shared object -> LDS base offset 0 (needed for the TDM descriptor).
  // layout: [buf][hi/lo][16 codes x KPAD]
  __shared__ __align__(16) unsigned short lds[2][2][CHUNK_ELTS];

  const int tid  = threadIdx.x;
  const int lane = tid & 31;
  const int wv   = tid >> 5;
  const int hihalf = lane >> 4;                 // 0: lanes 0-15, 1: lanes 16-31
  const int pix0 = blockIdx.x * PXT + wv * 16;
  const int mypix = pix0 + (lane & 15);
  const int b    = mypix >> 10;                 // batch (1024 px per image)
  const int poff = mypix & 1023;                // h*32+w

  // ===== build register-resident B fragments (pixels), bf16 hi/lo, K padded =====
  // B(32x16) layout: lane -> column N = lane&15; lanes<16 hold K {0..7,16..23} of the
  // 32-chunk, lanes>=16 hold K {8..15,24..31}; VGPR i packs K = k0 (lo16), k0+1 (hi16).
  v8u Bh[NKC], Bl[NKC];
  {
    const float* zb = z + (size_t)b * KDIM * 1024 + poff;  // channel stride 1024
    const int kb = hihalf ? 8 : 0;
#pragma unroll
    for (int j = 0; j < NKC; ++j) {
#pragma unroll
      for (int i = 0; i < 8; ++i) {
        int k0 = j * 32 + kb + 2 * i + ((i & 4) ? 8 : 0);
        int k1 = k0 + 1;
        float x0 = (k0 < KDIM) ? (-2.f * zb[(size_t)k0 * 1024]) : ((k0 == KDIM) ? 1.f : 0.f);
        float x1 = (k1 < KDIM) ? (-2.f * zb[(size_t)k1 * 1024]) : ((k1 == KDIM) ? 1.f : 0.f);
        unsigned h0, l0, h1, l1;
        split2(x0, h0, l0);
        split2(x1, h1, l1);
        Bh[j][i] = h0 | (h1 << 16);
        Bl[j][i] = l0 | (l1 << 16);
      }
    }
  }

  float rmin[8];
  int   ridx[8];
#pragma unroll
  for (int r = 0; r < 8; ++r) { rmin[r] = 3.4e38f; ridx[r] = 0; }

  // A(16x32) layout: lane -> row M = lane&15; lanes<16 hold K {0..7,16..23},
  // lanes>=16 hold K {8..15,24..31}; two ds_read_b128 per fragment.
  const int arow = (lane & 15) * KPAD;
  const int ko   = hihalf * 8;

  // A-fragment loader: two b128 LDS reads -> one v8u
  auto LDA = [&](const unsigned short* base, int j, v8u &dst) {
    uint4 p0 = *(const uint4*)(base + j * 32 + ko);
    uint4 p1 = *(const uint4*)(base + j * 32 + 16 + ko);
    dst[0] = p0.x; dst[1] = p0.y; dst[2] = p0.z; dst[3] = p0.w;
    dst[4] = p1.x; dst[5] = p1.y; dst[6] = p1.z; dst[7] = p1.w;
  };

  // LDS byte offsets of the four tiles (base 0)
  auto LDSOFF = [](int buf, int p) -> unsigned {
    return (unsigned)(buf * (2 * CHUNK_ELTS * 2) + p * (CHUNK_ELTS * 2));
  };

  // ===== TDM prologue: DMA chunk 0 into buffer 0 (wave 0 drives the DMA) =====
  if (wv == 0) {
    tdm_load_chunk(ehi, LDSOFF(0, 0));
    tdm_load_chunk(elo, LDSOFF(0, 1));
    __builtin_amdgcn_s_wait_tensorcnt(0);
  }
  __syncthreads();

  for (int c = 0; c < NCHUNK; ++c) {
    const int buf = c & 1;
    // kick off DMA of next chunk into the other buffer; runs concurrent with WMMAs
    if (wv == 0 && c + 1 < NCHUNK) {
      tdm_load_chunk(ehi + (size_t)(c + 1) * CHUNK_ELTS, LDSOFF(buf ^ 1, 0));
      tdm_load_chunk(elo + (size_t)(c + 1) * CHUNK_ELTS, LDSOFF(buf ^ 1, 1));
    }

    v8f acc = {0.f, 0.f, 0.f, 0.f, 0.f, 0.f, 0.f, 0.f};
    const unsigned short* lh = &lds[buf][0][arow];
    const unsigned short* ll = &lds[buf][1][arow];

    // software-pipelined A-fragment DS loads: double-buffered register sets so
    // the ds_read latency of step j+1 overlaps the 3 WMMAs of step j
    v8u ah[2], al[2];
    LDA(lh, 0, ah[0]);
    LDA(ll, 0, al[0]);
#pragma unroll
    for (int j = 0; j < NKC; ++j) {
      const int cu = j & 1;
      if (j + 1 < NKC) {
        LDA(lh, j + 1, ah[cu ^ 1]);
        LDA(ll, j + 1, al[cu ^ 1]);
      }
      v16bf Ah  = __builtin_bit_cast(v16bf, ah[cu]);
      v16bf Al  = __builtin_bit_cast(v16bf, al[cu]);
      v16bf Bhv = __builtin_bit_cast(v16bf, Bh[j]);
      v16bf Blv = __builtin_bit_cast(v16bf, Bl[j]);
      // hi*hi + hi*lo + lo*hi  ->  ~fp32-accurate  ||e||^2 - 2 z.e
      acc = __builtin_amdgcn_wmma_f32_16x16x32_bf16(false, Ah, false, Bhv, (short)0, acc, false, false);
      acc = __builtin_amdgcn_wmma_f32_16x16x32_bf16(false, Ah, false, Blv, (short)0, acc, false, false);
      acc = __builtin_amdgcn_wmma_f32_16x16x32_bf16(false, Al, false, Bhv, (short)0, acc, false, false);
    }

    // C layout: VGPR r holds code row r (lanes<16) / r+8 (lanes>=16), N = pixel lane
    const int cbase = c * 16 + hihalf * 8;
#pragma unroll
    for (int r = 0; r < 8; ++r) {
      float d = acc[r];
      bool lt = d < rmin[r];
      rmin[r] = lt ? d : rmin[r];
      ridx[r] = lt ? (cbase + r) : ridx[r];
    }

    // wave 0 confirms next chunk's DMA landed, then everyone syncs
    if (wv == 0) __builtin_amdgcn_s_wait_tensorcnt(0);
    __syncthreads();
  }

  // ===== per-lane then cross-half argmin (tie -> lowest index, like jnp.argmin) =====
  float m = rmin[0]; int mi = ridx[0];
#pragma unroll
  for (int r = 1; r < 8; ++r) {
    if (rmin[r] < m || (rmin[r] == m && ridx[r] < mi)) { m = rmin[r]; mi = ridx[r]; }
  }
  {
    float m2 = __shfl_xor(m, 16, 32);
    int   i2 = __shfl_xor(mi, 16, 32);
    if (m2 < m || (m2 == m && i2 < mi)) { m = m2; mi = i2; }
  }
  // lanes L and L+16 now agree on (m, mi) for pixel lane&15

  // ===== epilogue: z_q gather/store (coalesced NCHW), fused loss =====
  float* zq = out + 1;                          // z_q_out region
  const float* erow = emb + (size_t)mi * KDIM;
  const size_t obase = (size_t)b * KDIM * 1024 + poff;
  float lsum = 0.f;
#pragma unroll 4
  for (int t = 0; t < 128; ++t) {
    int ch = 2 * t + hihalf;                    // half-wave handles even/odd channels
    size_t off = obase + (size_t)ch * 1024;
    float e  = erow[ch];
    float zv = z[off];
    float dd = e - zv;
    lsum += dd * dd;
    zq[off] = e;                                // straight-through value == z_q
  }
#pragma unroll
  for (int off = 16; off >= 1; off >>= 1) lsum += __shfl_xor(lsum, off, 32);

  // block loss reduction: overlay scratch on the (now dead) codebook LDS
  float* lred = (float*)&lds[0][0][0];
  if (lane == 0) lred[wv] = lsum;
  __syncthreads();
  if (tid == 0) {
    float t = 0.f;
#pragma unroll
    for (int w = 0; w < 8; ++w) t += lred[w];
    loss_part[blockIdx.x] = t;                  // deterministic per-block partial
  }
  if (lane < 16) {
    atomicAdd(&counts[mi], 1.f);                // integer-valued adds: exact & order-free
    out[2 + ZQ_N + mypix] = (float)mi;          // idx output
  }
}

// ---------- kernel 3: loss scalar + perplexity ----------
__global__ void vq_fin(const float* __restrict__ counts,
                       const float* __restrict__ loss_part,
                       float* __restrict__ out) {
  __shared__ float red[8];
  int tid = threadIdx.x, lane = tid & 31, wv = tid >> 5;
  float s = 0.f;
  for (int i = tid; i < NE; i += 256) {
    float em = counts[i] * (1.f / (float)NPIX);
    s += em * logf(em + 1e-10f);
  }
#pragma unroll
  for (int off = 16; off >= 1; off >>= 1) s += __shfl_xor(s, off, 32);
  if (lane == 0) red[wv] = s;
  __syncthreads();
  if (tid == 0) {
    float t = 0.f;
#pragma unroll
    for (int w = 0; w < 8; ++w) t += red[w];
    out[1 + ZQ_N] = expf(-t);                   // perplexity
    float ls = 0.f;
    for (int i = 0; i < NPIX / PXT; ++i) ls += loss_part[i];  // fixed order
    // loss = mean + beta*mean = 1.25 * mean((z_q - z)^2)
    out[0] = 1.25f * ls / ((float)NPIX * (float)KDIM);
  }
}

extern "C" void kernel_launch(void* const* d_in, const int* in_sizes, int n_in,
                              void* d_out, int out_size, void* d_ws, size_t ws_size,
                              hipStream_t stream) {
  (void)in_sizes; (void)n_in; (void)out_size; (void)ws_size;
  const float* z   = (const float*)d_in[0];
  const float* emb = (const float*)d_in[1];

  // workspace layout: ehi | elo | counts[2048] | loss_part[256]
  unsigned short* ehi = (unsigned short*)d_ws;
  unsigned short* elo = ehi + (size_t)NE * KPAD;
  float* counts    = (float*)((char*)d_ws + (size_t)NE * KPAD * 2 * 2);
  float* loss_part = counts + NE;
  float* out = (float*)d_out;

  vq_zero<<<(NE + 255) / 256, 256, 0, stream>>>(counts);
  vq_prep<<<NE / 8, 256, 0, stream>>>(emb, ehi, elo);
  vq_main<<<NPIX / PXT, 256, 0, stream>>>(z, emb, ehi, elo, counts, loss_part, out);
  vq_fin<<<1, 256, 0, stream>>>(counts, loss_part, out);
}
